// GRUModel_56023553409425
// MI455X (gfx1250) — matrix operand order repaired and checked by reading, not compiled, not run on hardware
//
#include <hip/hip_runtime.h>
#include <stdint.h>

// GRU dims (fixed by the reference)
#define BATCH 4096
#define IN_C  8
#define SEQ_T 512
#define H1    32
#define H2    16
#define WAVES_PER_BLOCK 4

// x staging: chunks of 16 timesteps, double buffered per wave.
#define TCHUNK   16
#define NCHUNK   (SEQ_T / TCHUNK)          // 32
#define CHUNK_FLOATS (16 * IN_C * TCHUNK)  // 2048 floats = 8KB per buffer

typedef __attribute__((ext_vector_type(16))) _Float16 v16h;
typedef __attribute__((ext_vector_type(8)))  _Float16 v8h;
typedef __attribute__((ext_vector_type(8)))  float    v8f;

// In-wave LDS ordering fence: DS ops from one wave complete in order; we only
// need to wait for our own stores before cross-lane reads, and stop the
// compiler from reordering around it.
#define LDS_FENCE() asm volatile("s_wait_dscnt 0x0" ::: "memory")

__device__ __forceinline__ v8f splat8(float s) {
    v8f v;
#pragma unroll
    for (int e = 0; e < 8; ++e) v[e] = s;
    return v;
}
__device__ __forceinline__ v8f zero8f() { return splat8(0.0f); }
__device__ __forceinline__ v16h zero16h() {
    v16h v;
#pragma unroll
    for (int e = 0; e < 16; ++e) v[e] = (_Float16)0.0f;
    return v;
}

__device__ __forceinline__ float fsigmoid(float x) {
    float e = __expf(-x);
    return __builtin_amdgcn_rcpf(1.0f + e);
}
__device__ __forceinline__ float ftanh_(float x) {
    float e = __expf(2.0f * x);
    return (e - 1.0f) * __builtin_amdgcn_rcpf(e + 1.0f);
}

// Build one f16 B-operand tile (K=32, N=16) from row-major W[N_total][ldk].
// Contiguous k order used here (16-bit, 32x16): lanes 0-15 hold K=0..15 (2 per VGPR),
// lanes 16-31 hold K=16..31; lane%16 selects column N. K >= kvalid padded 0.
__device__ __forceinline__ v16h load_b_tile(const float* __restrict__ W, int ldk,
                                            int kvalid, int n0, int lcol, int half) {
    v16h b;
    const int n = n0 + lcol;
#pragma unroll
    for (int e = 0; e < 16; ++e) {
        const int k = half * 16 + e;
        const float f = (k < kvalid) ? W[n * ldk + k] : 0.0f;
        b[e] = (_Float16)f;
    }
    return b;
}

// Issue one x chunk (16 batch rows x 8 ch x TCHUNK t = 8KB) as 16 wave-wide
// async b128 copies into LDS. Granule g (16B) = (b, i, q) row-major, which is
// exactly linear in the LDS destination: lds = base + g*16.
// Global: row (b*8+i) has stride SEQ_T*4 bytes; q indexes 4 floats of t.
__device__ __forceinline__ void issue_x_chunk(uint32_t lds_dst_base,     // wave-relative LDS byte addr
                                              const float* __restrict__ xg,  // x + m0*IN_C*SEQ_T
                                              int chunk, int lane) {
    const uint64_t base  = (uint64_t)(uintptr_t)xg;
    const uint32_t tcoff = (uint32_t)chunk * (TCHUNK * 4);
#pragma unroll
    for (int j = 0; j < 16; ++j) {
        const uint32_t g    = (uint32_t)(j * 32 + lane);
        const uint32_t goff = (g >> 2) * (SEQ_T * 4) + (g & 3) * 16 + tcoff;
        const uint32_t vdst = lds_dst_base + g * 16;
        asm volatile("global_load_async_to_lds_b128 %0, %1, %2"
                     :: "v"(vdst), "v"(goff), "s"(base)
                     : "memory");
    }
}

__global__ void __launch_bounds__(32 * WAVES_PER_BLOCK, 1)
gru2_fused_wmma(const float* __restrict__ x,     // [B, I, T]
                const float* __restrict__ Wih1,  // [96, 8]
                const float* __restrict__ Whh1,  // [96, 32]
                const float* __restrict__ bih1,  // [96]
                const float* __restrict__ bhh1,  // [96]
                const float* __restrict__ Wih2,  // [48, 32]
                const float* __restrict__ Whh2,  // [48, 16]
                const float* __restrict__ bih2,  // [48]
                const float* __restrict__ bhh2,  // [48]
                const float* __restrict__ Wlin,  // [1, 16]
                const float* __restrict__ blin,  // [1]
                float* __restrict__ out)         // [B]
{
    const int lane = threadIdx.x & 31;
    const int w    = threadIdx.x >> 5;
    const int lcol = lane & 15;   // N column (B/C/D) or M row (A)
    const int half = lane >> 4;   // lane half selects M+8 / K range
    const int m0   = (blockIdx.x * WAVES_PER_BLOCK + w) * 16;  // batch base

    // Per-wave LDS: double-buffered x staging + transpose scratch.
    __shared__ __align__(16) float    sx [WAVES_PER_BLOCK][2][CHUNK_FLOATS];  // [b][i][t]
    __shared__ __align__(16) _Float16 sh1[WAVES_PER_BLOCK][16 * 32];
    __shared__ __align__(16) _Float16 sh2[WAVES_PER_BLOCK][16 * 16];
    __shared__ __align__(16) float    so [WAVES_PER_BLOCK][16 * 16];

    // ---- Weights as resident f16 B-operands (K zero-padded to 32) ----
    v16h Bhh1[6], Bxw1[6];
    v16h Bxw2[3], Bhh2t[3];
#pragma unroll
    for (int j = 0; j < 6; ++j) {
        Bhh1[j] = load_b_tile(Whh1, H1, H1, 16 * j, lcol, half);     // K=32
        Bxw1[j] = load_b_tile(Wih1, IN_C, IN_C, 16 * j, lcol, half); // K=8 padded
    }
#pragma unroll
    for (int j = 0; j < 3; ++j) {
        Bxw2[j]  = load_b_tile(Wih2, H1, H1, 16 * j, lcol, half);    // K=32
        Bhh2t[j] = load_b_tile(Whh2, H2, H2, 16 * j, lcol, half);    // K=16 padded
    }

    // Per-lane bias scalars (bias[n] splat across M in the C tile)
    float cbih1[6], cbhh1[6], cbih2[3], cbhh2[3];
#pragma unroll
    for (int j = 0; j < 6; ++j) {
        cbih1[j] = bih1[16 * j + lcol];
        cbhh1[j] = bhh1[16 * j + lcol];
    }
#pragma unroll
    for (int j = 0; j < 3; ++j) {
        cbih2[j] = bih2[16 * j + lcol];
        cbhh2[j] = bhh2[16 * j + lcol];
    }

    // Recurrent state: D-layout f32 (lane holds n=lcol(+16j), m = r + 8*half)
    v8f h1d[2];
    h1d[0] = zero8f();
    h1d[1] = zero8f();
    v8f h2d = zero8f();
    // A-layout f16 operands of the previous hidden states (start at 0)
    v16h ah1 = zero16h();
    v16h ah2 = zero16h();

    // Wave's batch strip base in x[B,I,T]
    const float* xg = x + (size_t)m0 * (IN_C * SEQ_T);
    // Wave-relative LDS byte addresses of the two staging buffers
    const uint32_t sx0 = (uint32_t)(uintptr_t)(void*)&sx[w][0][0];
    const uint32_t sx1 = (uint32_t)(uintptr_t)(void*)&sx[w][1][0];

    // Prime the pipeline: chunk 0 -> buffer 0.
    issue_x_chunk(sx0, xg, 0, lane);

    for (int c = 0; c < NCHUNK; ++c) {
        // Overlap: start the async copy of the next chunk into the other buffer.
        if (c + 1 < NCHUNK)
            issue_x_chunk(((c + 1) & 1) ? sx1 : sx0, xg, c + 1, lane);

        // Wait until chunk c's 16 async ops have landed (in-order completion);
        // chunk c+1's 16 may still be in flight.
        if (c + 1 < NCHUNK) asm volatile("s_wait_asynccnt 0x10" ::: "memory");
        else                asm volatile("s_wait_asynccnt 0x0"  ::: "memory");

        const float* xb = &sx[w][c & 1][lcol * (IN_C * TCHUNK)];  // this lane's [i][t] block

#pragma unroll 1
        for (int tl = 0; tl < TCHUNK; ++tl) {
            // ---- x_t as f16 A-operand: M=batch rows (lanes 0-15), K=i ----
            v16h ax = zero16h();
#pragma unroll
            for (int e = 0; e < IN_C; ++e) {
                const float f = xb[e * TCHUNK + tl];               // ds_load_b32
                ax[e] = (_Float16)(half == 0 ? f : 0.0f);          // half 1 rows = K pad
            }

            // ---- Layer 1: gh = h1@Whh1^T + bhh1 ; xw = x@Wih1^T + bih1 ----
            v8f gh[6], xw[6];
#pragma unroll
            for (int j = 0; j < 6; ++j) {
                gh[j] = __builtin_amdgcn_wmma_f32_16x16x32_f16(
                    false, ah1, false, Bhh1[j], (short)0, splat8(cbhh1[j]), false, false);
                xw[j] = __builtin_amdgcn_wmma_f32_16x16x32_f16(
                    false, ax, false, Bxw1[j], (short)0, splat8(cbih1[j]), false, false);
            }

            // ---- Layer 1 gates (tiles: 0-1 = r, 2-3 = z, 4-5 = n) ----
#pragma unroll
            for (int jj = 0; jj < 2; ++jj) {
#pragma unroll
                for (int e = 0; e < 8; ++e) {
                    const float rg = fsigmoid(xw[jj][e] + gh[jj][e]);
                    const float zg = fsigmoid(xw[jj + 2][e] + gh[jj + 2][e]);
                    const float ng = ftanh_(xw[jj + 4][e] + rg * gh[jj + 4][e]);
                    h1d[jj][e] = (1.0f - zg) * ng + zg * h1d[jj][e];
                }
            }

            // ---- Transpose h1 (D layout) -> ah1 (A layout) via LDS, f16 ----
            {
                _Float16* p1 = sh1[w];
#pragma unroll
                for (int jj = 0; jj < 2; ++jj)
#pragma unroll
                    for (int r = 0; r < 8; ++r)
                        p1[(r + 8 * half) * 32 + jj * 16 + lcol] = (_Float16)h1d[jj][r];
                LDS_FENCE();
                const v8h lo = *(const v8h*)(p1 + lcol * 32 + half * 16);      // K = half*16..+7
                const v8h hi = *(const v8h*)(p1 + lcol * 32 + half * 16 + 8);  // K = half*16+8..+15
#pragma unroll
                for (int e = 0; e < 8; ++e) {
                    ah1[e]     = lo[e];
                    ah1[e + 8] = hi[e];
                }
            }

            // ---- Layer 2: gh2 uses h2_{t-1} (old ah2); xw2 uses new h1 ----
            v8f g2[3], x2[3];
#pragma unroll
            for (int j = 0; j < 3; ++j) {
                g2[j] = __builtin_amdgcn_wmma_f32_16x16x32_f16(
                    false, ah2, false, Bhh2t[j], (short)0, splat8(cbhh2[j]), false, false);
                x2[j] = __builtin_amdgcn_wmma_f32_16x16x32_f16(
                    false, ah1, false, Bxw2[j], (short)0, splat8(cbih2[j]), false, false);
            }
#pragma unroll
            for (int e = 0; e < 8; ++e) {
                const float rg = fsigmoid(x2[0][e] + g2[0][e]);
                const float zg = fsigmoid(x2[1][e] + g2[1][e]);
                const float ng = ftanh_(x2[2][e] + rg * g2[2][e]);
                h2d[e] = (1.0f - zg) * ng + zg * h2d[e];
            }

            // ---- Transpose h2 -> ah2 (K padded 16->32) ----
            {
                _Float16* p2 = sh2[w];
#pragma unroll
                for (int r = 0; r < 8; ++r)
                    p2[(r + 8 * half) * 16 + lcol] = (_Float16)h2d[r];
                LDS_FENCE();
                const v8h lo2 = *(const v8h*)(p2 + lcol * 16);                 // K = 0..7
                const v8h hi2 = *(const v8h*)(p2 + lcol * 16 + 8);             // K = 8..15
#pragma unroll
                for (int e = 0; e < 8; ++e) {
                    ah2[e]     = half == 0 ? lo2[e] : (_Float16)0.0f;  // half 1: K >= 16 pad
                    ah2[e + 8] = half == 0 ? hi2[e] : (_Float16)0.0f;
                }
            }
        }
    }

    // ---- Final linear on last h2: out[m] = b + sum_n h2[m][n] * Wlin[n] ----
    {
        float* po = so[w];
#pragma unroll
        for (int r = 0; r < 8; ++r)
            po[(r + 8 * half) * 16 + lcol] = h2d[r];
        LDS_FENCE();
        if (lane < 16) {
            float acc = blin[0];
#pragma unroll
            for (int n = 0; n < 16; ++n) acc += po[lane * 16 + n] * Wlin[n];
            out[m0 + lane] = acc;
        }
    }
}

extern "C" void kernel_launch(void* const* d_in, const int* in_sizes, int n_in,
                              void* d_out, int out_size, void* d_ws, size_t ws_size,
                              hipStream_t stream) {
    (void)in_sizes; (void)n_in; (void)d_ws; (void)ws_size; (void)out_size;
    const float* x    = (const float*)d_in[0];
    const float* Wih1 = (const float*)d_in[1];
    const float* Whh1 = (const float*)d_in[2];
    const float* bih1 = (const float*)d_in[3];
    const float* bhh1 = (const float*)d_in[4];
    const float* Wih2 = (const float*)d_in[5];
    const float* Whh2 = (const float*)d_in[6];
    const float* bih2 = (const float*)d_in[7];
    const float* bhh2 = (const float*)d_in[8];
    const float* Wlin = (const float*)d_in[9];
    const float* blin = (const float*)d_in[10];
    float* out = (float*)d_out;

    const int blocks = BATCH / (16 * WAVES_PER_BLOCK);  // 64 blocks x 128 threads = 256 waves
    gru2_fused_wmma<<<blocks, 32 * WAVES_PER_BLOCK, 0, stream>>>(
        x, Wih1, Whh1, bih1, bhh1, Wih2, Whh2, bih2, bhh2, Wlin, blin, out);
}
